// MultiheadAttention_29592324669580
// MI455X (gfx1250) — compile-verified, matrix operand order checked by reading, not run on hardware
//
#include <hip/hip_runtime.h>
#include <hip/hip_bf16.h>

#define DMODEL 1024
#define HEADS 16
#define DHEAD 64
#define SEQ 2048
#define BATCH 2
#define NTOK (BATCH * SEQ)

typedef __bf16 v16bf __attribute__((ext_vector_type(16)));
typedef float  v8f   __attribute__((ext_vector_type(8)));
typedef float  v4f   __attribute__((ext_vector_type(4)));
typedef unsigned int v4u __attribute__((ext_vector_type(4)));
typedef unsigned int v8u __attribute__((ext_vector_type(8)));
typedef int v4i __attribute__((ext_vector_type(4)));

#define GLOBAL_AS __attribute__((address_space(1)))
#define LDS_AS    __attribute__((address_space(3)))

#if __has_builtin(__builtin_amdgcn_global_load_async_to_lds_b128)
#define HAVE_ASYNC_LDS 1
#else
#define HAVE_ASYNC_LDS 0
#endif

static __device__ __forceinline__ void wait_async0() {
#if HAVE_ASYNC_LDS
#if __has_builtin(__builtin_amdgcn_s_wait_asynccnt)
  __builtin_amdgcn_s_wait_asynccnt(0);
#else
  asm volatile("s_wait_asynccnt 0x0" ::: "memory");
#endif
#endif
}

static __device__ __forceinline__ unsigned short f2bf_bits(float f) {
  unsigned u = __builtin_bit_cast(unsigned, f);
  u += 0x7fffu + ((u >> 16) & 1u);   // round-to-nearest-even
  return (unsigned short)(u >> 16);
}
static __device__ __forceinline__ unsigned pk2bf(float a, float b) {
  return (unsigned)f2bf_bits(a) | ((unsigned)f2bf_bits(b) << 16);
}
static __device__ __forceinline__ v8f vzero8() {
  v8f z = {0.f, 0.f, 0.f, 0.f, 0.f, 0.f, 0.f, 0.f};
  return z;
}
// Assemble a 16-element bf16 WMMA operand from two 16-byte u32 loads with
// pure register coalescing (no per-element 16-bit moves).
static __device__ __forceinline__ v16bf pack16u(v4u g0, v4u g1) {
  v8u u = __builtin_shufflevector(g0, g1, 0, 1, 2, 3, 4, 5, 6, 7);
  return __builtin_bit_cast(v16bf, u);
}

// ---------------- weight fp32 -> bf16 conversion ----------------
__global__ void cvt_f32_bf16(const float* __restrict__ src,
                             unsigned short* __restrict__ dst, int n) {
  int i = (blockIdx.x * blockDim.x + threadIdx.x) * 4;
  if (i + 3 < n) {
    v4f v = *reinterpret_cast<const v4f*>(src + i);
    unsigned long long p = (unsigned long long)pk2bf(v[0], v[1])
                         | ((unsigned long long)pk2bf(v[2], v[3]) << 32);
    *reinterpret_cast<unsigned long long*>(dst + i) = p;
  }
}

// ---------------- GEMM: y = (x @ W^T + b) * scale ----------------
// One wave computes a 16x64 tile (A reused across 4 WMMAs per K-chunk).
template <bool IN_F32, bool OUT_F32>
__global__ __launch_bounds__(256) void gemm_bias(
    const void* __restrict__ xin, const unsigned short* __restrict__ wbf,
    const float* __restrict__ bias, void* __restrict__ yout, float scale) {
  const int lane = threadIdx.x & 31;
  const int wave = threadIdx.x >> 5;
  const int hi = lane >> 4;       // half-wave select (ISA 16-bit A/B layout)
  const int lr = lane & 15;
  const int koff = hi * 8;
  const int mbase = blockIdx.y * 128 + wave * 16;
  const int nbase = blockIdx.x * 64;
  const int arow = mbase + lr;

  v8f acc[4];
#pragma unroll
  for (int nt = 0; nt < 4; ++nt) acc[nt] = vzero8();

  const float* xrf = (const float*)xin + (size_t)arow * DMODEL;
  const unsigned short* xrb = (const unsigned short*)xin + (size_t)arow * DMODEL;

  for (int k0 = 0; k0 < DMODEL; k0 += 32) {
    v16bf a;
    if constexpr (IN_F32) {
      v4f f0 = *reinterpret_cast<const v4f*>(xrf + k0 + koff);
      v4f f1 = *reinterpret_cast<const v4f*>(xrf + k0 + koff + 4);
      v4f f2 = *reinterpret_cast<const v4f*>(xrf + k0 + 16 + koff);
      v4f f3 = *reinterpret_cast<const v4f*>(xrf + k0 + 16 + koff + 4);
      v8u au;
#pragma unroll
      for (int i = 0; i < 2; ++i) {
        au[i]     = pk2bf(f0[2 * i], f0[2 * i + 1]);
        au[i + 2] = pk2bf(f1[2 * i], f1[2 * i + 1]);
        au[i + 4] = pk2bf(f2[2 * i], f2[2 * i + 1]);
        au[i + 6] = pk2bf(f3[2 * i], f3[2 * i + 1]);
      }
      a = __builtin_bit_cast(v16bf, au);
    } else {
      v4u g0 = *reinterpret_cast<const v4u*>(xrb + k0 + koff);
      v4u g1 = *reinterpret_cast<const v4u*>(xrb + k0 + 16 + koff);
      a = pack16u(g0, g1);
    }
#pragma unroll
    for (int nt = 0; nt < 4; ++nt) {
      // B column n == W row n: contiguous loads land in the ISA B layout
      const unsigned short* wr = wbf + (size_t)(nbase + nt * 16 + lr) * DMODEL;
      v4u h0 = *reinterpret_cast<const v4u*>(wr + k0 + koff);
      v4u h1 = *reinterpret_cast<const v4u*>(wr + k0 + 16 + koff);
      v16bf b = pack16u(h0, h1);
      acc[nt] = __builtin_amdgcn_wmma_f32_16x16x32_bf16(false, a, false, b,
                                                        (short)0, acc[nt],
                                                        false, false);
    }
  }

#pragma unroll
  for (int nt = 0; nt < 4; ++nt) {
    const int col = nbase + nt * 16 + lr;
    const float bv = bias[col];
#pragma unroll
    for (int r = 0; r < 8; ++r) {
      const int orow = mbase + r + hi * 8;   // C layout: vgpr r -> row r / r+8
      const float val = (acc[nt][r] + bv) * scale;
      if constexpr (OUT_F32)
        ((float*)yout)[(size_t)orow * DMODEL + col] = val;
      else
        ((unsigned short*)yout)[(size_t)orow * DMODEL + col] = f2bf_bits(val);
    }
  }
}

// ---------------- flash attention (one head-slice per block.y) ----------------
// Double-buffered LDS staging: chunk i+1's async DMA overlaps chunk i's WMMAs.
__global__ __launch_bounds__(256) void attn_kernel(
    const unsigned short* __restrict__ Qh, const unsigned short* __restrict__ Kh,
    const unsigned short* __restrict__ Vh, unsigned short* __restrict__ ctx) {
  __shared__ unsigned short Ksm[2][32 * 64];  // K chunk, row-major [key][d]
  __shared__ unsigned short Vt[2][64 * 32];   // V chunk transposed [d][key]
  __shared__ unsigned short Psm[8 * 512];     // per-wave 16x32 P tile

  const int lane = threadIdx.x & 31;
  const int wave = threadIdx.x >> 5;
  const int hi = lane >> 4;
  const int lr = lane & 15;
  const int koff = hi * 8;
  const int b = blockIdx.y >> 4;
  const int h = blockIdx.y & 15;
  const int qbase = blockIdx.x * 128 + wave * 16;
  const size_t qoff = (size_t)(b * SEQ + qbase + lr) * DMODEL + h * DHEAD;

  const int t = threadIdx.x;
  const int kn = t >> 3, kp = t & 7;          // K staging coords
  const int vn = t & 31, vd = (t >> 5) * 8;   // V staging coords
  const unsigned short* kbase =
      Kh + (size_t)(b * SEQ + kn) * DMODEL + h * DHEAD + kp * 8;
  const unsigned short* vbase =
      Vh + (size_t)(b * SEQ + vn) * DMODEL + h * DHEAD + vd;

  // Stage one 32-key chunk into LDS buffer `bf` (K via async DMA, V transposed)
  auto stage = [&](int kv0, int bf) {
    const unsigned short* kptr = kbase + (size_t)kv0 * DMODEL;
    const unsigned short* vptr = vbase + (size_t)kv0 * DMODEL;
#if HAVE_ASYNC_LDS
    __builtin_amdgcn_global_load_async_to_lds_b128(
        (GLOBAL_AS v4i*)kptr, (LDS_AS v4i*)&Ksm[bf][kn * 64 + kp * 8], 0, 0);
#else
    *reinterpret_cast<v4u*>(&Ksm[bf][kn * 64 + kp * 8]) =
        *reinterpret_cast<const v4u*>(kptr);
#endif
    v4u vv = *reinterpret_cast<const v4u*>(vptr);
    const unsigned short* vs = reinterpret_cast<const unsigned short*>(&vv);
#pragma unroll
    for (int i = 0; i < 8; ++i) Vt[bf][(vd + i) * 32 + vn] = vs[i];
  };

  // Preload this wave's Q tile in A-operand layout (scale already folded in)
  v16bf qa[2];
#pragma unroll
  for (int c = 0; c < 2; ++c) {
    v4u g0 = *reinterpret_cast<const v4u*>(Qh + qoff + c * 32 + koff);
    v4u g1 = *reinterpret_cast<const v4u*>(Qh + qoff + c * 32 + 16 + koff);
    qa[c] = pack16u(g0, g1);
  }

  // B operand of all ones: one WMMA computes softmax row-sums on the matrix pipe
  v16bf vones;
  {
    v8u ou;
#pragma unroll
    for (int i = 0; i < 8; ++i) ou[i] = 0x3F803F80u;  // bf16 1.0 pair
    vones = __builtin_bit_cast(v16bf, ou);
  }

  v8f o[4];
#pragma unroll
  for (int c = 0; c < 4; ++c) o[c] = vzero8();
  float mrow[8], lrow[8];
#pragma unroll
  for (int j = 0; j < 8; ++j) { mrow[j] = -1.0e30f; lrow[j] = 0.0f; }

  // ---- pipeline prologue: stage chunk 0 ----
  stage(0, 0);
  wait_async0();
  __syncthreads();

  int buf = 0;
  for (int kv0 = 0; kv0 < SEQ; kv0 += 32) {
    // Kick off next chunk's DMA before computing on the current one.
    if (kv0 + 32 < SEQ) stage(kv0 + 32, buf ^ 1);

    // ---- S = Q . K^T : two 16x16 score tiles over 32 keys ----
    v8f s0 = vzero8(), s1 = vzero8();
#pragma unroll
    for (int tt = 0; tt < 2; ++tt) {
      v8f s = vzero8();
#pragma unroll
      for (int c = 0; c < 2; ++c) {
        v4u g0 = *reinterpret_cast<const v4u*>(
            &Ksm[buf][(tt * 16 + lr) * 64 + c * 32 + koff]);
        v4u g1 = *reinterpret_cast<const v4u*>(
            &Ksm[buf][(tt * 16 + lr) * 64 + c * 32 + 16 + koff]);
        v16bf kb = pack16u(g0, g1);
        s = __builtin_amdgcn_wmma_f32_16x16x32_bf16(false, qa[c], false, kb,
                                                    (short)0, s, false, false);
      }
      if (tt == 0) s0 = s; else s1 = s;
    }

    // ---- online softmax: row-max via lane shuffles (rows sit in 16-lane groups) ----
    float alpha[8];
#pragma unroll
    for (int j = 0; j < 8; ++j) {
      float tmax = fmaxf(s0[j], s1[j]);
#pragma unroll
      for (int m = 1; m <= 8; m <<= 1)
        tmax = fmaxf(tmax, __shfl_xor(tmax, m, 32));
      const float mn = fmaxf(mrow[j], tmax);
      alpha[j] = __expf(mrow[j] - mn);
      s0[j] = __expf(s0[j] - mn);
      s1[j] = __expf(s1[j] - mn);
      mrow[j] = mn;
    }
#pragma unroll
    for (int c = 0; c < 4; ++c)
#pragma unroll
      for (int j = 0; j < 8; ++j) o[c][j] *= alpha[j];

    // ---- P: C layout -> A layout via per-wave LDS round trip ----
    unsigned short* P = &Psm[wave * 512];
#pragma unroll
    for (int j = 0; j < 8; ++j) {
      const int prow = j + hi * 8;
      P[prow * 32 + lr]      = f2bf_bits(s0[j]);
      P[prow * 32 + 16 + lr] = f2bf_bits(s1[j]);
    }
    v4u pg0 = *reinterpret_cast<const v4u*>(&P[lr * 32 + koff]);
    v4u pg1 = *reinterpret_cast<const v4u*>(&P[lr * 32 + 16 + koff]);
    v16bf pa = pack16u(pg0, pg1);

    // ---- row-sum on the matrix pipe: ls[j] = sum_k P[row_j][k] ----
    v8f ls = __builtin_amdgcn_wmma_f32_16x16x32_bf16(false, pa, false, vones,
                                                     (short)0, vzero8(),
                                                     false, false);
#pragma unroll
    for (int j = 0; j < 8; ++j) lrow[j] = lrow[j] * alpha[j] + ls[j];

    // ---- O += P . V (B operand from transposed V in LDS) ----
#pragma unroll
    for (int c = 0; c < 4; ++c) {
      v4u g0 = *reinterpret_cast<const v4u*>(&Vt[buf][(c * 16 + lr) * 32 + koff]);
      v4u g1 = *reinterpret_cast<const v4u*>(&Vt[buf][(c * 16 + lr) * 32 + 16 + koff]);
      v16bf vb = pack16u(g0, g1);
      o[c] = __builtin_amdgcn_wmma_f32_16x16x32_bf16(false, pa, false, vb,
                                                     (short)0, o[c], false, false);
    }

    // ---- pipeline epilogue for this iteration: next chunk's DMA done, flip ----
    wait_async0();
    __syncthreads();
    buf ^= 1;
  }

  // ---- normalize and write context (bf16, [token][h*64+d] layout) ----
#pragma unroll
  for (int j = 0; j < 8; ++j) lrow[j] = 1.0f / lrow[j];
#pragma unroll
  for (int c = 0; c < 4; ++c)
#pragma unroll
    for (int j = 0; j < 8; ++j) {
      const float val = o[c][j] * lrow[j];
      const size_t tok = (size_t)(b * SEQ + qbase + j + hi * 8);
      ctx[tok * DMODEL + h * DHEAD + c * 16 + lr] = f2bf_bits(val);
    }
}

// ---------------- host launch ----------------
extern "C" void kernel_launch(void* const* d_in, const int* in_sizes, int n_in,
                              void* d_out, int out_size, void* d_ws, size_t ws_size,
                              hipStream_t stream) {
  (void)in_sizes; (void)n_in; (void)out_size; (void)ws_size;
  const float* q   = (const float*)d_in[0];
  const float* k   = (const float*)d_in[1];
  const float* v   = (const float*)d_in[2];
  const float* w_q = (const float*)d_in[3];
  const float* b_q = (const float*)d_in[4];
  const float* w_k = (const float*)d_in[5];
  const float* b_k = (const float*)d_in[6];
  const float* w_v = (const float*)d_in[7];
  const float* b_v = (const float*)d_in[8];
  const float* w_o = (const float*)d_in[9];
  const float* b_o = (const float*)d_in[10];

  char* ws = (char*)d_ws;
  unsigned short* wqb = (unsigned short*)(ws + (0ull  << 20));
  unsigned short* wkb = (unsigned short*)(ws + (2ull  << 20));
  unsigned short* wvb = (unsigned short*)(ws + (4ull  << 20));
  unsigned short* wob = (unsigned short*)(ws + (6ull  << 20));
  unsigned short* Qh  = (unsigned short*)(ws + (8ull  << 20));
  unsigned short* Kh  = (unsigned short*)(ws + (16ull << 20));
  unsigned short* Vh  = (unsigned short*)(ws + (24ull << 20));
  unsigned short* ctx = (unsigned short*)(ws + (32ull << 20));

  const int NW = DMODEL * DMODEL;
  dim3 cblk(256), cgrd(NW / (256 * 4));
  cvt_f32_bf16<<<cgrd, cblk, 0, stream>>>(w_q, wqb, NW);
  cvt_f32_bf16<<<cgrd, cblk, 0, stream>>>(w_k, wkb, NW);
  cvt_f32_bf16<<<cgrd, cblk, 0, stream>>>(w_v, wvb, NW);
  cvt_f32_bf16<<<cgrd, cblk, 0, stream>>>(w_o, wob, NW);

  dim3 gg(DMODEL / 64, NTOK / 128);
  const float qscale = 0.125f;  // 1/sqrt(DHEAD) folded into Q projection
  gemm_bias<true,  false><<<gg, 256, 0, stream>>>(q, wqb, b_q, Qh, qscale);
  gemm_bias<true,  false><<<gg, 256, 0, stream>>>(k, wkb, b_k, Kh, 1.0f);
  gemm_bias<true,  false><<<gg, 256, 0, stream>>>(v, wvb, b_v, Vh, 1.0f);

  attn_kernel<<<dim3(SEQ / 128, BATCH * HEADS), 256, 0, stream>>>(Qh, Kh, Vh, ctx);

  gemm_bias<false, true><<<gg, 256, 0, stream>>>(ctx, wob, b_o, d_out, 1.0f);
}
